// SpiralAware_CrossDeformAttn2D_75788992905313
// MI455X (gfx1250) — matrix operand order repaired and checked by reading, not compiled, not run on hardware
//
#include <hip/hip_runtime.h>
#include <hip/hip_bf16.h>

// ---------------------------------------------------------------------------
// Types for CDNA5 WMMA (wave32): v_wmma_f32_16x16x32_bf16
// ---------------------------------------------------------------------------
typedef __bf16 v16bf __attribute__((ext_vector_type(16)));
typedef float  v8f   __attribute__((ext_vector_type(8)));

union Frag {
  uint4 u[2];   // 32 bytes = 16 bf16
  v16bf v;
};

__device__ __forceinline__ unsigned short f32_to_bf16(float f) {
  unsigned int u = __float_as_uint(f);
  unsigned int lsb = (u >> 16) & 1u;
  u += 0x7fffu + lsb;                    // round to nearest even
  return (unsigned short)(u >> 16);
}

#define EPS 1e-5f

// ---------------------------------------------------------------------------
// Prep kernels
// ---------------------------------------------------------------------------

// Fold BN scale into conv weights, re-layout OIHW f32 -> [co][tap*256+ci] bf16
__global__ __launch_bounds__(256)
void fold_conv_w_kernel(const float* __restrict__ W, const float* __restrict__ g,
                        const float* __restrict__ var, unsigned short* __restrict__ Wt) {
  int idx = blockIdx.x * 256 + threadIdx.x;             // 256*2304 elements
  if (idx >= 256 * 2304) return;
  int co = idx / 2304;
  int k  = idx - co * 2304;
  int tap = k >> 8;            // k = tap*256 + ci
  int ci  = k & 255;
  float scale = g[co] * rsqrtf(var[co] + EPS);
  float w = W[(size_t)co * 2304 + (size_t)ci * 9 + tap]; // OIHW: [co][ci][kh][kw]
  Wt[(size_t)co * 2304 + k] = f32_to_bf16(w * scale);
}

__global__ __launch_bounds__(256)
void fold_conv_bias_kernel(const float* __restrict__ b, const float* __restrict__ g,
                           const float* __restrict__ beta, const float* __restrict__ mean,
                           const float* __restrict__ var, float* __restrict__ out) {
  int co = threadIdx.x;
  float scale = g[co] * rsqrtf(var[co] + EPS);
  out[co] = (b[co] - mean[co]) * scale + beta[co];
}

// [K][N] f32 -> [N][K] bf16 (N=K=256)
__global__ __launch_bounds__(256)
void transpose_bf16_kernel(const float* __restrict__ W, unsigned short* __restrict__ Wt) {
  int idx = blockIdx.x * 256 + threadIdx.x;             // 65536
  int n = idx >> 8;
  int k = idx & 255;
  Wt[idx] = f32_to_bf16(W[(size_t)k * 256 + n]);
}

// NCHW f32 -> NHWC bf16
__global__ __launch_bounds__(256)
void nchw2nhwc_kernel(const float* __restrict__ X, unsigned short* __restrict__ Y,
                      int HW, long total) {
  long idx = (long)blockIdx.x * 256 + threadIdx.x;
  if (idx >= total) return;
  long chw = (long)256 * HW;
  int b = (int)(idx / chw);
  long r = idx - (long)b * chw;
  int c = (int)(r / HW);
  int p = (int)(r - (long)c * HW);
  Y[((size_t)b * HW + p) * 256 + c] = f32_to_bf16(X[idx]);
}

// ---------------------------------------------------------------------------
// Implicit-GEMM 3x3 conv (pad 1) + folded-BN bias + ReLU, bf16 WMMA, f32 out.
// X:  [B, H*W, 256] bf16 (NHWC).  Wt: [256][9*256] bf16 (k = tap*256+ci).
// Block tile 128Mx128N, wave tile 32Mx64N, double-buffered LDS.
// Grid: (HW/128, 2, B), block 256.
// ---------------------------------------------------------------------------
__global__ __launch_bounds__(256)
void conv_wmma_kernel(const unsigned short* __restrict__ X,
                      const unsigned short* __restrict__ Wt,
                      const float* __restrict__ bias,
                      float* __restrict__ Y, int H, int Wd) {
  __shared__ unsigned short As[2][128 * 32];   // 2 x 8KB
  __shared__ unsigned short Bs[2][128 * 32];   // 2 x 8KB

  const int tid  = threadIdx.x;
  const int b    = blockIdx.z;
  const int m0   = blockIdx.x * 128;
  const int n0   = blockIdx.y * 128;
  const int lane = tid & 31, wave = tid >> 5;
  const int wm   = wave & 3, wn = wave >> 2;
  const int hi   = lane >> 4, lo = lane & 15;

  v8f acc[2][4];
  #pragma unroll
  for (int a = 0; a < 2; ++a)
    #pragma unroll
    for (int i = 0; i < 4; ++i)
      #pragma unroll
      for (int j = 0; j < 8; ++j) acc[a][i][j] = 0.0f;

  // fill mapping: each thread stages 32B (16 bf16) of one row for A and B
  const int frow = tid >> 1, fcol = (tid & 1) * 16;
  const int pix = m0 + frow;
  const int py = pix / Wd, px = pix - py * Wd;
  const unsigned short* wrow = Wt + (size_t)(n0 + frow) * 2304;
  const size_t xbase = (size_t)b * H * Wd;

  uint4 ra0, ra1, rb0, rb1;

  // fetch chunk kit into registers
  auto fetch = [&](int kit) {
    const int tap = kit >> 3;
    const int kc0 = (kit & 7) * 32;
    const int dy = tap / 3 - 1, dx = tap % 3 - 1;
    int yy = py + dy, xx = px + dx;
    ra0 = make_uint4(0u, 0u, 0u, 0u);
    ra1 = ra0;
    if (yy >= 0 && yy < H && xx >= 0 && xx < Wd) {
      const unsigned short* xp = X + (xbase + (size_t)yy * Wd + xx) * 256 + kc0 + fcol;
      ra0 = *(const uint4*)(xp);
      ra1 = *(const uint4*)(xp + 8);
    }
    const unsigned short* wp = wrow + tap * 256 + kc0 + fcol;
    rb0 = *(const uint4*)(wp);
    rb1 = *(const uint4*)(wp + 8);
  };
  auto stage = [&](int buf) {
    *(uint4*)(As[buf] + frow * 32 + fcol)     = ra0;
    *(uint4*)(As[buf] + frow * 32 + fcol + 8) = ra1;
    *(uint4*)(Bs[buf] + frow * 32 + fcol)     = rb0;
    *(uint4*)(Bs[buf] + frow * 32 + fcol + 8) = rb1;
  };

  fetch(0);
  stage(0);
  __syncthreads();

  const int KITERS = 72;
  for (int kit = 0; kit < KITERS; ++kit) {
    const int cur = kit & 1;
    if (kit + 1 < KITERS) fetch(kit + 1);    // overlap global latency with WMMA

    // A fragments: rows wm*32 + {0,16} + lo; elems 0..15=k[hi*16..hi*16+15]
    Frag af[2];
    #pragma unroll
    for (int a = 0; a < 2; ++a) {
      const unsigned short* ap = As[cur] + (wm * 32 + a * 16 + lo) * 32;
      af[a].u[0] = *(const uint4*)(ap + hi * 16);
      af[a].u[1] = *(const uint4*)(ap + hi * 16 + 8);
    }
    #pragma unroll
    for (int nt = 0; nt < 4; ++nt) {
      Frag bf;
      const unsigned short* bp = Bs[cur] + (wn * 64 + nt * 16 + lo) * 32 + hi * 16;
      bf.u[0] = *(const uint4*)(bp);
      bf.u[1] = *(const uint4*)(bp + 8);
      acc[0][nt] = __builtin_amdgcn_wmma_f32_16x16x32_bf16(
          false, af[0].v, false, bf.v, (short)0, acc[0][nt], false, false);
      acc[1][nt] = __builtin_amdgcn_wmma_f32_16x16x32_bf16(
          false, af[1].v, false, bf.v, (short)0, acc[1][nt], false, false);
    }
    if (kit + 1 < KITERS) stage(cur ^ 1);
    __syncthreads();   // publishes buf cur^1, retires reads of buf cur
  }

  // epilogue: bias + ReLU, f32 NHWC
  #pragma unroll
  for (int nt = 0; nt < 4; ++nt) {
    int n = n0 + wn * 64 + nt * 16 + lo;
    float bv = bias[n];
    #pragma unroll
    for (int a = 0; a < 2; ++a) {
      #pragma unroll
      for (int r = 0; r < 8; ++r) {
        int m = m0 + wm * 32 + a * 16 + hi * 8 + r;
        float v = fmaxf(acc[a][nt][r] + bv, 0.0f);
        Y[(xbase + m) * 256 + n] = v;
      }
    }
  }
}

// ---------------------------------------------------------------------------
// Dense GEMM: Y[M,256] = A[M,K]bf16 @ Bt[N,K]bf16^T + bias.
// Block tile 128x128, wave tile 32x64, double-buffered LDS.  Grid (M/128, 2).
// ---------------------------------------------------------------------------
__global__ __launch_bounds__(256)
void gemm_wmma_kernel(const unsigned short* __restrict__ A,
                      const unsigned short* __restrict__ Bt,
                      const float* __restrict__ bias,
                      float* __restrict__ Y, int K) {
  __shared__ unsigned short As[2][128 * 32];
  __shared__ unsigned short Bs[2][128 * 32];

  const int tid  = threadIdx.x;
  const int m0   = blockIdx.x * 128;
  const int n0   = blockIdx.y * 128;
  const int lane = tid & 31, wave = tid >> 5;
  const int wm   = wave & 3, wn = wave >> 2;
  const int hi   = lane >> 4, lo = lane & 15;

  v8f acc[2][4];
  #pragma unroll
  for (int a = 0; a < 2; ++a)
    #pragma unroll
    for (int i = 0; i < 4; ++i)
      #pragma unroll
      for (int j = 0; j < 8; ++j) acc[a][i][j] = 0.0f;

  const int frow = tid >> 1, fcol = (tid & 1) * 16;
  const unsigned short* arow = A + (size_t)(m0 + frow) * K + fcol;
  const unsigned short* brow = Bt + (size_t)(n0 + frow) * K + fcol;

  uint4 ra0, ra1, rb0, rb1;
  auto fetch = [&](int kit) {
    const int kc0 = kit * 32;
    ra0 = *(const uint4*)(arow + kc0);
    ra1 = *(const uint4*)(arow + kc0 + 8);
    rb0 = *(const uint4*)(brow + kc0);
    rb1 = *(const uint4*)(brow + kc0 + 8);
  };
  auto stage = [&](int buf) {
    *(uint4*)(As[buf] + frow * 32 + fcol)     = ra0;
    *(uint4*)(As[buf] + frow * 32 + fcol + 8) = ra1;
    *(uint4*)(Bs[buf] + frow * 32 + fcol)     = rb0;
    *(uint4*)(Bs[buf] + frow * 32 + fcol + 8) = rb1;
  };

  fetch(0);
  stage(0);
  __syncthreads();

  const int kiters = K >> 5;
  for (int kit = 0; kit < kiters; ++kit) {
    const int cur = kit & 1;
    if (kit + 1 < kiters) fetch(kit + 1);

    Frag af[2];
    #pragma unroll
    for (int a = 0; a < 2; ++a) {
      const unsigned short* ap = As[cur] + (wm * 32 + a * 16 + lo) * 32;
      af[a].u[0] = *(const uint4*)(ap + hi * 16);
      af[a].u[1] = *(const uint4*)(ap + hi * 16 + 8);
    }
    #pragma unroll
    for (int nt = 0; nt < 4; ++nt) {
      Frag bf;
      const unsigned short* bp = Bs[cur] + (wn * 64 + nt * 16 + lo) * 32 + hi * 16;
      bf.u[0] = *(const uint4*)(bp);
      bf.u[1] = *(const uint4*)(bp + 8);
      acc[0][nt] = __builtin_amdgcn_wmma_f32_16x16x32_bf16(
          false, af[0].v, false, bf.v, (short)0, acc[0][nt], false, false);
      acc[1][nt] = __builtin_amdgcn_wmma_f32_16x16x32_bf16(
          false, af[1].v, false, bf.v, (short)0, acc[1][nt], false, false);
    }
    if (kit + 1 < kiters) stage(cur ^ 1);
    __syncthreads();
  }

  #pragma unroll
  for (int nt = 0; nt < 4; ++nt) {
    int n = n0 + wn * 64 + nt * 16 + lo;
    float bv = bias[n];
    #pragma unroll
    for (int a = 0; a < 2; ++a) {
      #pragma unroll
      for (int r = 0; r < 8; ++r) {
        int m = m0 + wm * 32 + a * 16 + hi * 8 + r;
        Y[(size_t)m * 256 + n] = acc[a][nt][r] + bv;
      }
    }
  }
}

// ---------------------------------------------------------------------------
// LayerNorm over C=256; one wave per row; writes f32 (optional) + bf16.
// ---------------------------------------------------------------------------
__global__ __launch_bounds__(256)
void ln_kernel(const float* __restrict__ X, const float* __restrict__ g,
               const float* __restrict__ bta, float* __restrict__ Yf,
               unsigned short* __restrict__ Yb, int writeF) {
  int wave = threadIdx.x >> 5, lane = threadIdx.x & 31;
  size_t row = (size_t)blockIdx.x * 8 + wave;
  const float* xr = X + row * 256;
  float x[8], s = 0.0f;
  #pragma unroll
  for (int i = 0; i < 8; ++i) { x[i] = xr[lane + 32 * i]; s += x[i]; }
  #pragma unroll
  for (int m = 16; m >= 1; m >>= 1) s += __shfl_xor(s, m, 32);
  float mu = s * (1.0f / 256.0f);
  float vs = 0.0f;
  #pragma unroll
  for (int i = 0; i < 8; ++i) { float d = x[i] - mu; vs += d * d; }
  #pragma unroll
  for (int m = 16; m >= 1; m >>= 1) vs += __shfl_xor(vs, m, 32);
  float rs = rsqrtf(vs * (1.0f / 256.0f) + EPS);
  #pragma unroll
  for (int i = 0; i < 8; ++i) {
    int c = lane + 32 * i;
    float yv = (x[i] - mu) * rs * g[c] + bta[c];
    if (writeF) Yf[row * 256 + c] = yv;
    Yb[row * 256 + c] = f32_to_bf16(yv);
  }
}

// ---------------------------------------------------------------------------
// Attention weights: scores = Q @ awW + awB; softmax over each group of 4.
// One wave per query row, lane j = h*4+p (32 outputs).
// ---------------------------------------------------------------------------
__global__ __launch_bounds__(256)
void aw_softmax_kernel(const float* __restrict__ Q, const float* __restrict__ awW,
                       const float* __restrict__ awB, float* __restrict__ AW) {
  __shared__ float qs[8][256];
  int wave = threadIdx.x >> 5, lane = threadIdx.x & 31;
  size_t row = (size_t)blockIdx.x * 8 + wave;
  #pragma unroll
  for (int i = 0; i < 8; ++i)
    qs[wave][lane + 32 * i] = Q[row * 256 + lane + 32 * i];
  __syncthreads();
  float s = awB[lane];
  for (int c = 0; c < 256; ++c) s += qs[wave][c] * awW[c * 32 + lane];
  float m = fmaxf(s, __shfl_xor(s, 1, 32));
  m = fmaxf(m, __shfl_xor(m, 2, 32));
  float e = expf(s - m);
  float d = e + __shfl_xor(e, 1, 32);
  d += __shfl_xor(d, 2, 32);
  AW[row * 32 + lane] = e / d;
}

// ---------------------------------------------------------------------------
// Bilinear sample + weighted aggregation.  One wave per (b,h,q); lane = d.
// value: [B,1024,256] f32 (head h at cols h*32..).  Output AGG bf16 [B,4096,256].
// ---------------------------------------------------------------------------
__global__ __launch_bounds__(256)
void sample_agg_kernel(const float* __restrict__ value, const float* __restrict__ AW,
                       const float* __restrict__ off_res, unsigned short* __restrict__ AGG) {
  int wv   = blockIdx.x * 8 + (threadIdx.x >> 5);
  int lane = threadIdx.x & 31;
  int q = wv & 4095;
  int h = (wv >> 12) & 7;
  int b = wv >> 15;

  float refx = ((q & 63) + 0.5f) * (1.0f / 64.0f);
  float refy = ((q >> 6) + 0.5f) * (1.0f / 64.0f);
  const float* vbase = value + (size_t)b * 1024 * 256 + h * 32 + lane;
  const float* awrow = AW + ((size_t)b * 4096 + q) * 32 + h * 4;

  float acc = 0.0f;
  #pragma unroll
  for (int p = 0; p < 4; ++p) {
    float ang = 6.283185307179586f * (0.25f * p + 0.125f * h);
    float rad = 1.0f + (float)p;
    float ox = rad * cosf(ang) + off_res[(h * 4 + p) * 2 + 0];
    float oy = rad * sinf(ang) + off_res[(h * 4 + p) * 2 + 1];
    float x = refx * 32.0f + ox - 0.5f;
    float y = refy * 32.0f + oy - 0.5f;
    float x0f = floorf(x), y0f = floorf(y);
    float fx = x - x0f, fy = y - y0f;
    int x0 = (int)x0f, y0 = (int)y0f;
    int x1 = x0 + 1, y1 = y0 + 1;
    float a = awrow[p];
    float w00 = a * (1.0f - fx) * (1.0f - fy);
    float w10 = a * fx * (1.0f - fy);
    float w01 = a * (1.0f - fx) * fy;
    float w11 = a * fx * fy;
    if (x0 >= 0 && x0 < 32 && y0 >= 0 && y0 < 32) acc += w00 * vbase[(size_t)(y0 * 32 + x0) * 256];
    if (x1 >= 0 && x1 < 32 && y0 >= 0 && y0 < 32) acc += w10 * vbase[(size_t)(y0 * 32 + x1) * 256];
    if (x0 >= 0 && x0 < 32 && y1 >= 0 && y1 < 32) acc += w01 * vbase[(size_t)(y1 * 32 + x0) * 256];
    if (x1 >= 0 && x1 < 32 && y1 >= 0 && y1 < 32) acc += w11 * vbase[(size_t)(y1 * 32 + x1) * 256];
  }
  AGG[((size_t)b * 4096 + q) * 256 + h * 32 + lane] = f32_to_bf16(acc);
}

// ---------------------------------------------------------------------------
// Final: t = q + q*attn; LayerNorm; write NCHW f32.
// ---------------------------------------------------------------------------
__global__ __launch_bounds__(256)
void final_ln_kernel(const float* __restrict__ Q, const float* __restrict__ Ain,
                     const float* __restrict__ g, const float* __restrict__ bta,
                     float* __restrict__ OUT) {
  int wave = threadIdx.x >> 5, lane = threadIdx.x & 31;
  size_t row = (size_t)blockIdx.x * 8 + wave;      // 32768 rows
  int b  = (int)(row >> 12);
  int qq = (int)(row & 4095);
  float x[8], s = 0.0f;
  #pragma unroll
  for (int i = 0; i < 8; ++i) {
    int c = lane + 32 * i;
    float qv = Q[row * 256 + c];
    float av = Ain[row * 256 + c];
    x[i] = qv + qv * av;
    s += x[i];
  }
  #pragma unroll
  for (int m = 16; m >= 1; m >>= 1) s += __shfl_xor(s, m, 32);
  float mu = s * (1.0f / 256.0f);
  float vs = 0.0f;
  #pragma unroll
  for (int i = 0; i < 8; ++i) { float d = x[i] - mu; vs += d * d; }
  #pragma unroll
  for (int m = 16; m >= 1; m >>= 1) vs += __shfl_xor(vs, m, 32);
  float rs = rsqrtf(vs * (1.0f / 256.0f) + EPS);
  #pragma unroll
  for (int i = 0; i < 8; ++i) {
    int c = lane + 32 * i;
    float yv = (x[i] - mu) * rs * g[c] + bta[c];
    OUT[((size_t)b * 256 + c) * 4096 + qq] = yv;
  }
}

// ---------------------------------------------------------------------------
extern "C" void kernel_launch(void* const* d_in, const int* in_sizes, int n_in,
                              void* d_out, int out_size, void* d_ws, size_t ws_size,
                              hipStream_t stream) {
  (void)in_sizes; (void)n_in; (void)out_size; (void)ws_size;

  const float* query_feat = (const float*)d_in[0];
  const float* key_feat   = (const float*)d_in[1];
  const float* qc_w  = (const float*)d_in[2];
  const float* qc_b  = (const float*)d_in[3];
  const float* qbn_g = (const float*)d_in[4];
  const float* qbn_b = (const float*)d_in[5];
  const float* qbn_m = (const float*)d_in[6];
  const float* qbn_v = (const float*)d_in[7];
  const float* kc_w  = (const float*)d_in[8];
  const float* kc_b  = (const float*)d_in[9];
  const float* kbn_g = (const float*)d_in[10];
  const float* kbn_b = (const float*)d_in[11];
  const float* kbn_m = (const float*)d_in[12];
  const float* kbn_v = (const float*)d_in[13];
  const float* off_res = (const float*)d_in[14];
  const float* lnq_g = (const float*)d_in[15];
  const float* lnq_b = (const float*)d_in[16];
  const float* lnk_g = (const float*)d_in[17];
  const float* lnk_b = (const float*)d_in[18];
  const float* lno_g = (const float*)d_in[19];
  const float* lno_b = (const float*)d_in[20];
  const float* vproj_w = (const float*)d_in[21];
  const float* vproj_b = (const float*)d_in[22];
  const float* aw_w    = (const float*)d_in[23];
  const float* aw_b    = (const float*)d_in[24];
  const float* oproj_w = (const float*)d_in[25];
  const float* oproj_b = (const float*)d_in[26];

  char* ws = (char*)d_ws;
  const size_t OFF_WQT   = 0;                            // 256*2304*2
  const size_t OFF_WKT   = OFF_WQT + 1179648;
  const size_t OFF_BIASQ = OFF_WKT + 1179648;            // 1 KB
  const size_t OFF_BIASK = OFF_BIASQ + 1024;
  const size_t OFF_VPW   = OFF_BIASK + 1024;             // 128 KB
  const size_t OFF_OPW   = OFF_VPW + 131072;
  const size_t OFF_XQ    = OFF_OPW + 131072;             // 16 MB  (query bf16 NHWC)
  const size_t OFF_XK    = OFF_XQ + 16777216;            // 4 MB
  const size_t OFF_QF    = OFF_XK + 4194304;             // 32 MB  (conv-q out; reused as attn_out)
  const size_t OFF_KF    = OFF_QF + 33554432;            // 8 MB   (conv-k out; reused as value)
  const size_t OFF_QRY   = OFF_KF + 8388608;             // 32 MB  (query post-LN f32)
  const size_t OFF_QRYB  = OFF_QRY + 33554432;           // 16 MB  (query post-LN bf16)
  const size_t OFF_KVB   = OFF_QRYB + 16777216;          // 4 MB   (kv post-LN bf16)
  const size_t OFF_AW    = OFF_KVB + 4194304;            // 4 MB
  const size_t OFF_AGG   = OFF_AW + 4194304;             // 16 MB

  unsigned short* wqt  = (unsigned short*)(ws + OFF_WQT);
  unsigned short* wkt  = (unsigned short*)(ws + OFF_WKT);
  float* biasq         = (float*)(ws + OFF_BIASQ);
  float* biask         = (float*)(ws + OFF_BIASK);
  unsigned short* vpw  = (unsigned short*)(ws + OFF_VPW);
  unsigned short* opw  = (unsigned short*)(ws + OFF_OPW);
  unsigned short* xq   = (unsigned short*)(ws + OFF_XQ);
  unsigned short* xk   = (unsigned short*)(ws + OFF_XK);
  float* qf            = (float*)(ws + OFF_QF);
  float* kf            = (float*)(ws + OFF_KF);
  float* qry           = (float*)(ws + OFF_QRY);
  unsigned short* qryb = (unsigned short*)(ws + OFF_QRYB);
  unsigned short* kvb  = (unsigned short*)(ws + OFF_KVB);
  float* awbuf         = (float*)(ws + OFF_AW);
  unsigned short* aggb = (unsigned short*)(ws + OFF_AGG);
  float* value         = kf;   // reuse after LN(kv)
  float* attn          = qf;   // reuse after LN(q)

  // 1. weight prep
  fold_conv_w_kernel<<<2304, 256, 0, stream>>>(qc_w, qbn_g, qbn_v, wqt);
  fold_conv_bias_kernel<<<1, 256, 0, stream>>>(qc_b, qbn_g, qbn_b, qbn_m, qbn_v, biasq);
  fold_conv_w_kernel<<<2304, 256, 0, stream>>>(kc_w, kbn_g, kbn_v, wkt);
  fold_conv_bias_kernel<<<1, 256, 0, stream>>>(kc_b, kbn_g, kbn_b, kbn_m, kbn_v, biask);
  transpose_bf16_kernel<<<256, 256, 0, stream>>>(vproj_w, vpw);
  transpose_bf16_kernel<<<256, 256, 0, stream>>>(oproj_w, opw);

  // 2. activation layout conversion
  nchw2nhwc_kernel<<<32768, 256, 0, stream>>>(query_feat, xq, 4096, 8388608L);
  nchw2nhwc_kernel<<<8192, 256, 0, stream>>>(key_feat, xk, 1024, 2097152L);

  // 3. convs (WMMA implicit GEMM) + BN + ReLU
  conv_wmma_kernel<<<dim3(32, 2, 8), 256, 0, stream>>>(xq, wqt, biasq, qf, 64, 64);
  conv_wmma_kernel<<<dim3(8, 2, 8), 256, 0, stream>>>(xk, wkt, biask, kf, 32, 32);

  // 4. layernorms
  ln_kernel<<<4096, 256, 0, stream>>>(qf, lnq_g, lnq_b, qry, qryb, 1);
  ln_kernel<<<1024, 256, 0, stream>>>(kf, lnk_g, lnk_b, nullptr, kvb, 0);

  // 5. value projection (WMMA GEMM), writes into kf (free after LN)
  gemm_wmma_kernel<<<dim3(64, 2), 256, 0, stream>>>(kvb, vpw, vproj_b, value, 256);

  // 6. attention weights + softmax
  aw_softmax_kernel<<<4096, 256, 0, stream>>>(qry, aw_w, aw_b, awbuf);

  // 7. deformable bilinear sampling + aggregation -> bf16 agg
  sample_agg_kernel<<<32768, 256, 0, stream>>>(value, awbuf, off_res, aggb);

  // 8. output projection (WMMA GEMM), writes into qf (free after LN)
  gemm_wmma_kernel<<<dim3(256, 2), 256, 0, stream>>>(aggb, opw, oproj_b, attn, 256);

  // 9. residual gate + final LN + NCHW transpose
  final_ln_kernel<<<4096, 256, 0, stream>>>(qry, attn, lno_g, lno_b, (float*)d_out);
}
